// QuantumLayer_69690139345157
// MI455X (gfx1250) — compile-verified, matrix operand order checked by reading, not run on hardware
//
#include <hip/hip_runtime.h>
#include <hip/hip_bf16.h>

// ---------------------------------------------------------------------------
// QuantumLayer for MI455X (gfx1250):
//   amp  = clip(x,-10,10) @ W_in^T + b_in           (WMMA f16->f32 GEMM)
//   state= normalize(amp) as complex                 (LDS-resident)
//   6 layers of 12 RZ*RY*RX gates + CNOT ring        (LDS butterflies/swaps)
//   out  = (|state|^2 @ SIGNS^T) @ W_out^T + b_out   (sign reductions)
// ---------------------------------------------------------------------------

#define GM   8192      // batch
#define GK   1024      // input dim
#define GN   4096      // state dim (2^12)
#define NQ   12
#define NLAY 6
#define ODIM 16
#define ROWS 4         // state rows per sim workgroup

typedef __attribute__((ext_vector_type(16))) _Float16 v16h;
typedef __attribute__((ext_vector_type(8)))  float    v8f;

// ---------------------------------------------------------------------------
// f32 -> f16 conversion (x gets the reference's clip to [-10,10])
// ---------------------------------------------------------------------------
__global__ void cvt_x_f16(const float* __restrict__ x, _Float16* __restrict__ xh, int n) {
  for (int i = blockIdx.x * blockDim.x + threadIdx.x; i < n; i += gridDim.x * blockDim.x) {
    float v = x[i];
    v = fminf(10.0f, fmaxf(-10.0f, v));
    xh[i] = (_Float16)v;
  }
}

__global__ void cvt_w_f16(const float* __restrict__ w, _Float16* __restrict__ wh, int n) {
  for (int i = blockIdx.x * blockDim.x + threadIdx.x; i < n; i += gridDim.x * blockDim.x) {
    wh[i] = (_Float16)w[i];
  }
}

// ---------------------------------------------------------------------------
// Fused single-qubit unitaries: U = RZ(tz) * RY(ty) * RX(tx), 2x2 complex.
// Ubuf[(l*12+q)*8] = {u00r,u00i,u01r,u01i,u10r,u10i,u11r,u11i}
// ---------------------------------------------------------------------------
__global__ void prep_u(const float* __restrict__ qp, float* __restrict__ Ubuf) {
  int id = blockIdx.x * blockDim.x + threadIdx.x;
  if (id >= NLAY * NQ) return;
  float tx = qp[id * 3 + 0], ty = qp[id * 3 + 1], tz = qp[id * 3 + 2];
  float cx = cosf(tx * 0.5f), sx = sinf(tx * 0.5f);
  float cy = cosf(ty * 0.5f), sy = sinf(ty * 0.5f);
  float er = cosf(tz * 0.5f), ei = -sinf(tz * 0.5f);   // e^{-i tz/2}
  // M = RY @ RX
  float m00r =  cy * cx, m00i =  sy * sx;
  float m01r = -sy * cx, m01i = -cy * sx;
  float m10r =  sy * cx, m10i = -cy * sx;
  float m11r =  cy * cx, m11i = -sy * sx;
  float* U = Ubuf + (size_t)id * 8;
  // row 0: ez * M[0][:]
  U[0] = er * m00r - ei * m00i;  U[1] = er * m00i + ei * m00r;
  U[2] = er * m01r - ei * m01i;  U[3] = er * m01i + ei * m01r;
  // row 1: conj(ez) * M[1][:]
  U[4] = er * m10r + ei * m10i;  U[5] = er * m10i - ei * m10r;
  U[6] = er * m11r + ei * m11i;  U[7] = er * m11i - ei * m11r;
}

// ---------------------------------------------------------------------------
// WMMA GEMM: amp[M][N] = Xh[M][K] * Wh[N][K]^T + b_in[N]
// One wave computes a 64x32 tile with 4x2 register blocking of 16x16x32
// fragments: 12 b128 loads feed 8 WMMAs per K-step (1.5 loads/WMMA vs 2.0
// for 2x2), cutting L1/L2 fetch per FLOP by ~1.3x and giving 8 independent
// accumulator chains for latency hiding.
// Fragment layout per CDNA5 ISA (16-bit A, 16x32):
//   lane r = lane&15 -> matrix row; half = lane>>4 selects K sub-pattern:
//   halves 0..7  = K[kb + half*8 .. +8), halves 8..15 = K[kb+16+half*8 .. +8)
// Both operands loaded K-major with the same pattern -> D = A * B^T.
// ---------------------------------------------------------------------------
__device__ __forceinline__ v16h load_frag(const _Float16* __restrict__ rowp, int k, int half) {
  union { v16h v; uint4 q[2]; } u;
  u.q[0] = *(const uint4*)(rowp + k + half * 8);
  u.q[1] = *(const uint4*)(rowp + k + 16 + half * 8);
  return u.v;
}

__global__ __launch_bounds__(256)
void gemm_wmma(const _Float16* __restrict__ Xh,
               const _Float16* __restrict__ Wh,
               const float* __restrict__ b_in,
               float* __restrict__ amp) {
  const int lane = threadIdx.x & 31;
  const int wave = blockIdx.x * (blockDim.x >> 5) + (threadIdx.x >> 5);
  const int NT = GN / 32;                 // 128 tiles along N
  const int mt = wave / NT;               // 0..127  (64-row M tiles)
  const int nt = wave % NT;               // 0..127  (32-col N tiles)
  const int r    = lane & 15;
  const int half = lane >> 4;

  const _Float16* xp = Xh + (size_t)(mt * 64 + r) * GK;
  const _Float16* wp = Wh + (size_t)(nt * 32 + r) * GK;

  v8f acc[4][2];
#pragma unroll
  for (int mi = 0; mi < 4; ++mi)
#pragma unroll
    for (int ni = 0; ni < 2; ++ni) acc[mi][ni] = (v8f){};

  for (int k = 0; k < GK; k += 32) {
    v16h fb0 = load_frag(wp, k, half);
    v16h fb1 = load_frag(wp + (size_t)16 * GK, k, half);
    v16h fa0 = load_frag(xp, k, half);
    v16h fa1 = load_frag(xp + (size_t)16 * GK, k, half);
    v16h fa2 = load_frag(xp + (size_t)32 * GK, k, half);
    v16h fa3 = load_frag(xp + (size_t)48 * GK, k, half);
    acc[0][0] = __builtin_amdgcn_wmma_f32_16x16x32_f16(false, fa0, false, fb0, (short)0, acc[0][0], false, false);
    acc[0][1] = __builtin_amdgcn_wmma_f32_16x16x32_f16(false, fa0, false, fb1, (short)0, acc[0][1], false, false);
    acc[1][0] = __builtin_amdgcn_wmma_f32_16x16x32_f16(false, fa1, false, fb0, (short)0, acc[1][0], false, false);
    acc[1][1] = __builtin_amdgcn_wmma_f32_16x16x32_f16(false, fa1, false, fb1, (short)0, acc[1][1], false, false);
    acc[2][0] = __builtin_amdgcn_wmma_f32_16x16x32_f16(false, fa2, false, fb0, (short)0, acc[2][0], false, false);
    acc[2][1] = __builtin_amdgcn_wmma_f32_16x16x32_f16(false, fa2, false, fb1, (short)0, acc[2][1], false, false);
    acc[3][0] = __builtin_amdgcn_wmma_f32_16x16x32_f16(false, fa3, false, fb0, (short)0, acc[3][0], false, false);
    acc[3][1] = __builtin_amdgcn_wmma_f32_16x16x32_f16(false, fa3, false, fb1, (short)0, acc[3][1], false, false);
  }

  // C/D layout: element (M = mBase + mi*16 + v, N = nBase + ni*16)
  const int nBase = nt * 32 + (lane & 15);
  const int mBase = mt * 64 + (lane >> 4) * 8;
  const float bi0 = b_in[nBase];
  const float bi1 = b_in[nBase + 16];
#pragma unroll
  for (int mi = 0; mi < 4; ++mi) {
#pragma unroll
    for (int v = 0; v < 8; ++v) {
      const size_t mrow = (size_t)(mBase + mi * 16 + v) * GN;
      amp[mrow + nBase]      = acc[mi][0][v] + bi0;
      amp[mrow + nBase + 16] = acc[mi][1][v] + bi1;
    }
  }
}

// ---------------------------------------------------------------------------
// Quantum circuit simulation. 256 threads = 4 groups of 64; each group owns
// one state vector (4096 complex f32) resident in LDS (16KB re + 16KB im).
// Dynamic LDS: 4 rows -> ~132KB (2 workgroups per 320KB WGP).
// ---------------------------------------------------------------------------
__global__ void quantum_sim(const float* __restrict__ amp,
                            const float* __restrict__ Ubuf,
                            const float* __restrict__ W_out,
                            const float* __restrict__ b_out,
                            float* __restrict__ out) {
  extern __shared__ float smem[];
  float* SR  = smem;                     // ROWS*GN
  float* SI  = SR + ROWS * GN;           // ROWS*GN
  float* RED = SI + ROWS * GN;           // ROWS*64
  float* EZS = RED + ROWS * 64;          // ROWS*NQ

  const int g   = threadIdx.x >> 6;
  const int t   = threadIdx.x & 63;
  const int row = blockIdx.x * ROWS + g;
  float* re  = SR  + g * GN;
  float* im  = SI  + g * GN;
  float* red = RED + g * 64;
  float* ez  = EZS + g * NQ;

  // ---- load amp row, compute L2 norm, normalize (reference semantics) ----
  const float* arow = amp + (size_t)row * GN;
  float ss = 0.0f;
#pragma unroll
  for (int j = 0; j < GN / 64; ++j) {
    int k = t + j * 64;
    float v = arow[k];
    re[k] = v;
    ss += v * v;
  }
  red[t] = ss;
  __syncthreads();
#pragma unroll
  for (int s = 32; s > 0; s >>= 1) {
    if (t < s) red[t] += red[t + s];
    __syncthreads();
  }
  const float nrm = sqrtf(red[0]);
  const bool  ok  = nrm > 1e-8f;
  const float scl = ok ? 1.0f / fmaxf(nrm, 1e-8f) : 0.0f;
  __syncthreads();
#pragma unroll
  for (int j = 0; j < GN / 64; ++j) {
    int k = t + j * 64;
    re[k] = ok ? re[k] * scl : (k == 0 ? 1.0f : 0.0f);
    im[k] = 0.0f;
  }
  __syncthreads();

  // ---- 6 variational layers ----
  for (int l = 0; l < NLAY; ++l) {
    // 12 single-qubit gates: butterfly on pairs with stride 2^(11-q)
    for (int q = 0; q < NQ; ++q) {
      const float* U = Ubuf + (size_t)(l * NQ + q) * 8;
      const float u00r = U[0], u00i = U[1], u01r = U[2], u01i = U[3];
      const float u10r = U[4], u10i = U[5], u11r = U[6], u11i = U[7];
      const int s = 1 << (11 - q);
#pragma unroll 4
      for (int p = t; p < GN / 2; p += 64) {
        int i0 = ((p & ~(s - 1)) << 1) | (p & (s - 1));
        int i1 = i0 + s;
        float ar = re[i0], ai = im[i0];
        float br = re[i1], bi = im[i1];
        re[i0] = u00r * ar - u00i * ai + u01r * br - u01i * bi;
        im[i0] = u00r * ai + u00i * ar + u01r * bi + u01i * br;
        re[i1] = u10r * ar - u10i * ai + u11r * br - u11i * bi;
        im[i1] = u10r * ai + u10i * ar + u11r * bi + u11i * br;
      }
      __syncthreads();
    }
    // CNOT ring: (c, c+1) for c=0..10, then (11, 0). Pure in-place swaps:
    // swap indices with bit[cb]=1, bit[tb]=0 <-> bit[tb]=1.
    for (int c = 0; c < NQ; ++c) {
      const int tq = (c + 1) % NQ;
      const int cb = 11 - c, tb = 11 - tq;
      const int hi = cb > tb ? cb : tb;
      const int lo = cb ^ tb ^ hi;
      const int midw = hi - lo - 1;
#pragma unroll 4
      for (int p = t; p < GN / 4; p += 64) {
        unsigned v   = (unsigned)p;
        unsigned low = v & ((1u << lo) - 1u);
        v >>= lo;
        unsigned mid = v & ((1u << midw) - 1u);
        v >>= midw;
        unsigned i0 = (v << (hi + 1)) | (mid << (lo + 1)) | low | (1u << cb);
        unsigned i1 = i0 | (1u << tb);
        float tr = re[i0], ti = im[i0];
        re[i0] = re[i1];  im[i0] = im[i1];
        re[i1] = tr;      im[i1] = ti;
      }
      __syncthreads();
    }
  }

  // ---- expz[i] = sum_k |psi_k|^2 * (1 - 2*bit(k, 11-i)) ----
  float part[NQ];
#pragma unroll
  for (int i = 0; i < NQ; ++i) part[i] = 0.0f;
#pragma unroll 2
  for (int j = 0; j < GN / 64; ++j) {
    int k = t + j * 64;
    float pr = re[k] * re[k] + im[k] * im[k];
#pragma unroll
    for (int i = 0; i < NQ; ++i) {
      part[i] += ((k >> (11 - i)) & 1) ? -pr : pr;
    }
  }
  for (int i = 0; i < NQ; ++i) {
    red[t] = part[i];
    __syncthreads();
#pragma unroll
    for (int s = 32; s > 0; s >>= 1) {
      if (t < s) red[t] += red[t + s];
      __syncthreads();
    }
    if (t == 0) ez[i] = red[0];
    __syncthreads();
  }

  // ---- out[row][o] = sum_i ez[i] * W_out[o][i] + b_out[o] ----
  if (t < ODIM) {
    float acc = b_out[t];
#pragma unroll
    for (int i = 0; i < NQ; ++i) acc += ez[i] * W_out[t * NQ + i];
    out[(size_t)row * ODIM + t] = acc;
  }
}

// ---------------------------------------------------------------------------
// Host side
// ---------------------------------------------------------------------------
extern "C" void kernel_launch(void* const* d_in, const int* in_sizes, int n_in,
                              void* d_out, int out_size, void* d_ws, size_t ws_size,
                              hipStream_t stream) {
  (void)in_sizes; (void)n_in; (void)out_size; (void)ws_size;
  const float* x        = (const float*)d_in[0];   // [8192,1024]
  const float* W_in     = (const float*)d_in[1];   // [4096,1024]
  const float* b_in     = (const float*)d_in[2];   // [4096]
  const float* q_params = (const float*)d_in[3];   // [216]
  const float* W_out    = (const float*)d_in[4];   // [16,12]
  const float* b_out    = (const float*)d_in[5];   // [16]
  float* out = (float*)d_out;                      // [8192,16]

  // Workspace layout (~152 MB):
  char* ws = (char*)d_ws;
  _Float16* Xh = (_Float16*)ws;                                // 16 MB
  _Float16* Wh = Xh + (size_t)GM * GK;                         //  8 MB
  float* Ubuf  = (float*)(Wh + (size_t)GN * GK);               // 2.3 KB
  char* ampRaw = (char*)(Ubuf + NLAY * NQ * 8);
  ampRaw = (char*)(((uintptr_t)ampRaw + 255) & ~(uintptr_t)255);
  float* amp   = (float*)ampRaw;                               // 128 MB

  // 1) conversions (x gets the reference clip)
  cvt_x_f16<<<4096, 256, 0, stream>>>(x, Xh, GM * GK);
  cvt_w_f16<<<4096, 256, 0, stream>>>(W_in, Wh, GN * GK);

  // 2) fused per-qubit unitaries
  prep_u<<<1, 96, 0, stream>>>(q_params, Ubuf);

  // 3) input GEMM via WMMA: (8192/64)*(4096/32) = 16384 waves, 8 waves/block
  gemm_wmma<<<2048, 256, 0, stream>>>(Xh, Wh, b_in, amp);

  // 4) LDS-resident quantum circuit + readout
  const int smemBytes = (ROWS * GN * 2 + ROWS * 64 + ROWS * NQ) * (int)sizeof(float);
  quantum_sim<<<GM / ROWS, 256, smemBytes, stream>>>(amp, Ubuf, W_out, b_out, out);
}
